// VanillaLSTM_20615843020861
// MI455X (gfx1250) — matrix operand order repaired and checked by reading, not compiled, not run on hardware
//
#include <hip/hip_runtime.h>

typedef __attribute__((ext_vector_type(16))) _Float16 v16h;
typedef __attribute__((ext_vector_type(8)))  float    v8f;

#define T_STEPS 64
#define NSEQ    131072
#define HID     20
#define MDIM    10
#define GDIM    80      // 4*H
#define KC      32      // padded K = MDIM + HID + 2 pad
#define WAVES   8
#define SPW     16      // sequences per wave (WMMA M dim)
#define BLOCKT  (WAVES * 32)
#define SEQ_PER_BLOCK (WAVES * SPW)   // 128

// gfx1250 has a native v_tanh_f32 transcendental -- use it for both tanh and
// sigmoid (sig(x) = 0.5*tanh(x/2) + 0.5) to avoid the branchy libm expansion.
#if defined(__has_builtin)
#if __has_builtin(__builtin_amdgcn_tanhf)
#define FAST_TANH(x) __builtin_amdgcn_tanhf(x)
#endif
#endif
#ifndef FAST_TANH
#define FAST_TANH(x) tanhf(x)
#endif

__device__ __forceinline__ float fast_tanh(float x) { return FAST_TANH(x); }
__device__ __forceinline__ float sigf(float x) {
    return fmaf(0.5f, fast_tanh(0.5f * x), 0.5f);
}

__global__ __launch_bounds__(BLOCKT)
void VanillaLSTM_wmma_kernel(
    const float* __restrict__ X,       // [T,N,4]
    const int*   __restrict__ masks,   // [T,1,N]
    const float* __restrict__ h0,      // [N,H]
    const float* __restrict__ c0,      // [N,H]
    const float* __restrict__ Y,       // [T,N,2]
    const int*   __restrict__ pTobs,
    const int*   __restrict__ pTpred,
    const float* __restrict__ W_in,    // [M,2]
    const float* __restrict__ b_in,    // [M]
    const float* __restrict__ W_ih,    // [4H,M]
    const float* __restrict__ W_hh,    // [4H,H]
    const float* __restrict__ b_ih,    // [4H]
    const float* __restrict__ b_hh,    // [4H]
    const float* __restrict__ W_out,   // [2,H]
    const float* __restrict__ b_out,   // [2]
    float*       __restrict__ out)     // [T,N,2]
{
    // Combined weight Wc[g][k]: k<10 -> W_ih[g][k]; 10<=k<30 -> W_hh[g][k-10]; else 0
    __shared__ __align__(16) _Float16 sWcB[GDIM][KC];          // B operand source, row = gate
    __shared__ __align__(16) _Float16 sA[WAVES][SPW][KC];      // per-wave [r | h | pad] rows
    __shared__ float sG[WAVES][SPW][GDIM];                     // gate pre-activations
    __shared__ float sBias[GDIM];
    __shared__ float sWin[MDIM * 2], sbin[MDIM];
    __shared__ float sWout[2 * HID], sbout[2];

    const int tid  = threadIdx.x;
    const int wave = tid >> 5;
    const int lane = tid & 31;
    const int s    = lane & 15;     // sequence-in-tile / matrix column index
    const int half = lane >> 4;
    const int n    = blockIdx.x * SEQ_PER_BLOCK + wave * SPW + s;

    // ---- one-time weight staging into LDS (fp16 for WMMA A/B) ----
    for (int idx = tid; idx < GDIM * KC; idx += BLOCKT) {
        const int g = idx / KC, k = idx % KC;
        float v = 0.0f;
        if (k < MDIM)            v = W_ih[g * MDIM + k];
        else if (k < MDIM + HID) v = W_hh[g * HID + (k - MDIM)];
        sWcB[g][k] = (_Float16)v;
    }
    if (tid < GDIM)     sBias[tid] = b_ih[tid] + b_hh[tid];
    if (tid < MDIM * 2) sWin[tid]  = W_in[tid];
    if (tid < MDIM)     sbin[tid]  = b_in[tid];
    if (tid < 2 * HID)  sWout[tid] = W_out[tid];
    if (tid < 2)        sbout[tid] = b_out[tid];

    // ---- initial state: h -> sA rows, c -> registers (lane = (seq, half)) ----
    float creg[10];
#pragma unroll
    for (int u = 0; u < 10; ++u) {
        const int hu = half * 10 + u;
        creg[u] = c0[n * HID + hu];
        sA[wave][s][MDIM + hu] = (_Float16)h0[n * HID + hu];
    }
    if (half == 0) { sA[wave][s][30] = (_Float16)0.0f; sA[wave][s][31] = (_Float16)0.0f; }

    const int T_obs  = pTobs[0];
    const int T_pred = pTpred[0];

    __syncthreads();

    // ---- B fragments (time-invariant). Contiguous k order (k = 16*half + e):
    //      lanes 0-15: K=0..15 of column N=lane; lanes 16-31: K=16..31.
    v16h bf[5];
    {
        const int kb = half * 16;
#pragma unroll
        for (int j = 0; j < 5; ++j) {
            const _Float16* row = &sWcB[j * 16 + s][kb];
#pragma unroll
            for (int e = 0; e < 16; ++e) bf[j][e] = row[e];
        }
    }

    float po0 = 0.0f, po1 = 0.0f;       // prev output (carry)
    int m1 = 1, m2 = 1, m3 = 1;         // mask history t-1, t-2, t-3

    for (int t = 0; t < T_STEPS; ++t) {
        const int mt = masks[t * NSEQ + n];

        // phase a: lanes 0-15 compute r = relu(W_in * inp + b_in) -> sA[s][0..9]
        if (half == 0) {
            float i0, i1;
            if (t <= T_obs) {
                const float2 xv = *(const float2*)&X[(size_t)(t * NSEQ + n) * 4 + 2];
                i0 = xv.x; i1 = xv.y;
            } else { i0 = po0; i1 = po1; }
#pragma unroll
            for (int m = 0; m < MDIM; ++m) {
                const float r = fmaf(sWin[m * 2 + 1], i1, fmaf(sWin[m * 2], i0, sbin[m]));
                sA[wave][s][m] = (_Float16)fmaxf(r, 0.0f);
            }
        }
        __syncthreads();

        // A fragment: contiguous k order, same as B (row M = s):
        //   lanes 0-15: K in {0..15}; lanes 16-31: K in {16..31}
        v16h af;
        {
            const int kb = half * 16;
            const _Float16* row = &sA[wave][s][0];
#pragma unroll
            for (int e = 0; e < 8; ++e) af[e] = row[kb + e];
#pragma unroll
            for (int e = 0; e < 8; ++e) af[8 + e] = row[kb + 8 + e];
        }

        // gates tile GEMM: [16 seq x 32] x [32 x 80] via 5 WMMAs, dump C to LDS
#pragma unroll
        for (int j = 0; j < 5; ++j) {
            v8f c = {};
            c = __builtin_amdgcn_wmma_f32_16x16x32_f16(
                    false, af, false, bf[j], (short)0, c, false, false);
            const int row0 = half * 8;   // C layout: lanes<16 -> M=r, lanes>=16 -> M=8+r
#pragma unroll
            for (int r = 0; r < 8; ++r)
                sG[wave][row0 + r][j * 16 + s] = c[r];
        }
        __syncthreads();

        // phase d: LSTM cell nonlinearity + output projection (lane = (seq s, half))
        const bool active = (t <= T_pred);
        float p0 = 0.0f, p1 = 0.0f;
#pragma unroll
        for (int u = 0; u < 10; ++u) {
            const int hu = half * 10 + u;
            const float gi = sG[wave][s][hu]        + sBias[hu];
            const float gf = sG[wave][s][20 + hu]   + sBias[20 + hu];
            const float gg = sG[wave][s][40 + hu]   + sBias[40 + hu];
            const float go = sG[wave][s][60 + hu]   + sBias[60 + hu];
            const float cn = sigf(gf) * creg[u] + sigf(gi) * fast_tanh(gg);
            const float hn = sigf(go) * fast_tanh(cn);
            if (active) {
                creg[u] = cn;
                sA[wave][s][MDIM + hu] = (_Float16)hn;   // h for next step's A
            }
            p0 = fmaf(sWout[hu],       hn, p0);
            p1 = fmaf(sWout[HID + hu], hn, p1);
        }
        // reduce the two half-lane partials (xor across half boundary, wave32)
        p0 += __shfl_xor(p0, 16, 32);
        p1 += __shfl_xor(p1, 16, 32);

        float o0 = (p0 + sbout[0]) * (float)mt;
        float o1 = (p1 + sbout[1]) * (float)mt;

        const int  mp3 = (t < 3) ? 1 : m3;
        const bool ow  = (t > 3) && (t > T_obs) && (mt != 0) && (mp3 == 0);
        if (ow) {
            const float2 yv = *(const float2*)&Y[(size_t)(t * NSEQ + n) * 2];
            o0 = yv.x; o1 = yv.y;
        }
        if (!active) { o0 = 0.0f; o1 = 0.0f; }
        po0 = o0; po1 = o1;

        if (half == 0) {
            *(float2*)&out[(size_t)(t * NSEQ + n) * 2] = make_float2(o0, o1);
        }
        m3 = m2; m2 = m1; m1 = mt;
    }
}

extern "C" void kernel_launch(void* const* d_in, const int* in_sizes, int n_in,
                              void* d_out, int out_size, void* d_ws, size_t ws_size,
                              hipStream_t stream) {
    (void)in_sizes; (void)n_in; (void)out_size; (void)d_ws; (void)ws_size;
    const float* X      = (const float*)d_in[0];
    const int*   masks  = (const int*)  d_in[1];
    const float* h0     = (const float*)d_in[2];
    const float* c0     = (const float*)d_in[3];
    const float* Y      = (const float*)d_in[4];
    const int*   pTobs  = (const int*)  d_in[5];
    const int*   pTpred = (const int*)  d_in[6];
    const float* W_in   = (const float*)d_in[7];
    const float* b_in   = (const float*)d_in[8];
    const float* W_ih   = (const float*)d_in[9];
    const float* W_hh   = (const float*)d_in[10];
    const float* b_ih   = (const float*)d_in[11];
    const float* b_hh   = (const float*)d_in[12];
    const float* W_out  = (const float*)d_in[13];
    const float* b_out  = (const float*)d_in[14];
    float*       outp   = (float*)d_out;

    const int grid = NSEQ / SEQ_PER_BLOCK;   // 1024 blocks of 256 threads
    VanillaLSTM_wmma_kernel<<<grid, BLOCKT, 0, stream>>>(
        X, masks, h0, c0, Y, pTobs, pTpred,
        W_in, b_in, W_ih, W_hh, b_ih, b_hh, W_out, b_out, outp);
}
